// MultiHeadedSelfAttentionModule_30605936951352
// MI455X (gfx1250) — matrix operand order repaired and checked by reading, not compiled, not run on hardware
//
#include <hip/hip_runtime.h>

// ---------------------------------------------------------------------------
// MI455X / gfx1250 Transformer-XL multi-head self-attention (bf16 WMMA).
//   B=8, T=1024, D=512, H=8, DH=64.
// - All matrix math: v_wmma_f32_16x16x32_bf16 (fp32 accumulate).
// - Tile staging: global_load_async_to_lds_b128 (ASYNCcnt DMA) with a
//   double-buffered pipeline in the GEMM; V/weight tiles (which need a
//   transpose scatter) use the manual load+ds_store path.
// - Flash (online) softmax: the 256MB score tensor is never materialized;
//   only the 32MB per-batch rel-position matrix PS is, because the
//   reference's rel_shift (incl. its unmasked wrap values) is an exact
//   gather from PS.
// - Working set (~89MB) fits in the 192MB global L2, so after first touch
//   everything streams at L2 rates.
// ---------------------------------------------------------------------------

typedef __attribute__((ext_vector_type(16))) __bf16 bf16x16;
typedef __attribute__((ext_vector_type(8)))  float  f32x8;

#define DIMB 8
#define DIMT 1024
#define DIMD 512
#define DIMH 8
#define DIMDH 64
#define INV_SQRT_D 0.04419417382415922f   // 1/sqrt(512)
#define ATS 72   // padded LDS row stride (bf16): mult of 8, conflict-free for 16 rows

union FragU { bf16x16 v; unsigned int u[8]; };

__device__ __forceinline__ unsigned short f2bf(float x) {
  unsigned int u = __float_as_uint(x);
  unsigned int r = u + 0x7FFFu + ((u >> 16) & 1u);
  return (unsigned short)(r >> 16);
}
__device__ __forceinline__ f32x8 zero8() {
  f32x8 z;
#pragma unroll
  for (int i = 0; i < 8; ++i) z[i] = 0.0f;
  return z;
}

// ---- CDNA5 async global->LDS DMA (ASYNCcnt) -------------------------------
// the LDS address is the vdst VGPR plus the instruction offset, and that offset is also
// added to the global address (ISA §15.18 async pseudocode), so a 32B/lane
// transfer is two b128 ops sharing one address pair with offset:0 / offset:16.
// LDS offset = low 32 bits of the generic pointer (flat->local = truncate).
__device__ __forceinline__ void async_ld_b128x2(void* ldsDst, const void* gsrc) {
  unsigned lds = (unsigned)(uintptr_t)ldsDst;
  unsigned long long ga = (unsigned long long)(uintptr_t)gsrc;
  asm volatile("global_load_async_to_lds_b128 %0, %1, off\n\t"
               "global_load_async_to_lds_b128 %0, %1, off offset:16"
               :: "v"(lds), "v"(ga) : "memory");
}
template <int N>
__device__ __forceinline__ void wait_async() {
  asm volatile("s_wait_asynccnt %0" :: "n"(N) : "memory");
}

// A fragment (16x32 bf16): lane half=lane>>4, row m=lane&15.
// VGPR v holds K=half*16+2v (+1). rowPtr = 32-bit view of a
// contiguous bf16 row; koffU=16 selects K=32..63 of a 64-wide row.
__device__ __forceinline__ bf16x16 load_fragA(const unsigned int* rowPtr, int lane, int koffU) {
  int half = (lane >> 4) & 1;
  FragU f;
#pragma unroll
  for (int v = 0; v < 8; ++v)
    f.u[v] = rowPtr[koffU + half * 8 + v];
  return f.v;
}
// B fragment (32x16 bf16): lane half selects K 0..15/16..31, col n=lane&15.
// VGPR v holds K = half*16 + 2v (+1). rowPtr = row of B^T (contiguous in K).
__device__ __forceinline__ bf16x16 load_fragB(const unsigned int* rowPtr, int lane, int koffU) {
  int half = (lane >> 4) & 1;
  FragU f;
#pragma unroll
  for (int v = 0; v < 8; ++v)
    f.u[v] = rowPtr[koffU + half * 8 + v];
  return f.v;
}
__device__ __forceinline__ f32x8 wmma_bf16(bf16x16 a, bf16x16 b, f32x8 c) {
  return __builtin_amdgcn_wmma_f32_16x16x32_bf16(false, a, false, b, (short)0, c, false, false);
}

// ------------------------------ LayerNorm ----------------------------------
__global__ void __launch_bounds__(128)
ln_kernel(const float* __restrict__ x, const float* __restrict__ g,
          const float* __restrict__ bta, unsigned short* __restrict__ xn) {
  __shared__ float red[128], red2[128];
  int tok = blockIdx.x;
  int tid = threadIdx.x;
  const float* row = x + (size_t)tok * DIMD;
  float4 val = ((const float4*)row)[tid];
  float s  = val.x + val.y + val.z + val.w;
  float s2 = val.x * val.x + val.y * val.y + val.z * val.z + val.w * val.w;
  red[tid] = s; red2[tid] = s2;
  __syncthreads();
  for (int off = 64; off > 0; off >>= 1) {
    if (tid < off) { red[tid] += red[tid + off]; red2[tid] += red2[tid + off]; }
    __syncthreads();
  }
  float mean = red[0] * (1.0f / DIMD);
  float var  = red2[0] * (1.0f / DIMD) - mean * mean;
  float rstd = rsqrtf(var + 1e-5f);
  float4 gg = ((const float4*)g)[tid];
  float4 bb = ((const float4*)bta)[tid];
  union { unsigned short h[4]; unsigned int u[2]; } o;
  o.h[0] = f2bf((val.x - mean) * rstd * gg.x + bb.x);
  o.h[1] = f2bf((val.y - mean) * rstd * gg.y + bb.y);
  o.h[2] = f2bf((val.z - mean) * rstd * gg.z + bb.z);
  o.h[3] = f2bf((val.w - mean) * rstd * gg.w + bb.w);
  *(unsigned int*)(xn + (size_t)tok * DIMD + tid * 4 + 0) = o.u[0];
  *(unsigned int*)(xn + (size_t)tok * DIMD + tid * 4 + 2) = o.u[1];
}

// ------------------------- fp32 -> bf16 convert ----------------------------
__global__ void __launch_bounds__(256)
cvt_kernel(const float* __restrict__ in, unsigned short* __restrict__ out, int n) {
  int idx = blockIdx.x * 256 + threadIdx.x;
  if (idx < n) out[idx] = f2bf(in[idx]);
}

// ----------------------- sinusoidal positional enc -------------------------
__global__ void __launch_bounds__(256)
pe_kernel(unsigned short* __restrict__ pe) {
  int idx = blockIdx.x * 256 + threadIdx.x;        // < T*D
  int t = idx >> 9;
  int d = idx & (DIMD - 1);
  int i2 = d & ~1;                                 // 2*i
  float freq = __expf(-(float)i2 * (9.210340371976184f / (float)DIMD));
  float ang  = (float)t * freq;
  float vv   = (d & 1) ? __cosf(ang) : __sinf(ang);
  pe[idx] = f2bf(vv);
}

// ------------------- bias prep: bq+u_flat and bq+v_flat --------------------
__global__ void __launch_bounds__(256)
biasprep_kernel(const float* __restrict__ bq, const float* __restrict__ u,
                const float* __restrict__ vb, float* __restrict__ bqu,
                float* __restrict__ bqv) {
  int idx = blockIdx.x * 256 + threadIdx.x;
  if (idx < DIMD) { bqu[idx] = bq[idx] + u[idx]; bqv[idx] = bq[idx] + vb[idx]; }
}

// ------------------------- generic bf16 WMMA GEMM --------------------------
// C[M,N] = A[M,K]*B[K,N] (+bias). 64x64 block tile, BK=64, 256 threads
// (8 waves, 2 C subtiles each, 4 WMMAs per stage). Double-buffered LDS fed by
// global_load_async_to_lds_b128. BTRANS=1: B element (k,n)=Bsrc[(n0+n)*ldb+bcol+k]
// (async path); BTRANS=0: weights [K,N] row-major, manual scatter into B^T.
// z axis gives per-head slicing (acol+=z*za, bcol+=z*zb, C+=z*zc).
template <int BTRANS>
__global__ void __launch_bounds__(256)
gemm_wmma_bf16(const unsigned short* __restrict__ A, int lda, int acol, int za,
               const unsigned short* __restrict__ B, int ldb, int bcol, int zb,
               void* C0, void* C1, int out_f32, int ldc, long long zc,
               const float* __restrict__ bias0, const float* __restrict__ bias1,
               int M, int N, int K) {
  __shared__ unsigned short As[2][64 * ATS];
  __shared__ unsigned short Bt[2][64 * ATS];   // stored as B^T: [n][k]
  int tid = threadIdx.x, wave = tid >> 5, lane = tid & 31;
  int z = blockIdx.z;
  int m0 = blockIdx.y * 64, n0 = blockIdx.x * 64;
  acol += z * za; bcol += z * zb;
  size_t cbase = (size_t)z * (size_t)zc;
  int i0 = wave >> 1;
  int jb = (wave & 1) * 2;
  int r  = tid >> 2;           // 0..63
  int cg = (tid & 3) * 16;     // 0,16,32,48
  f32x8 acc0 = zero8(), acc1 = zero8();

  const int S = K >> 6;        // stages of BK=64
  const int PEND = BTRANS ? 4 : 2;  // async instrs per wave per stage

  // stage loader: async A tile (and async B^T tile if BTRANS)
  auto load_stage = [&](int buf, int k0) {
    const unsigned short* ga = A + (size_t)(m0 + r) * lda + acol + k0 + cg;
    async_ld_b128x2(&As[buf][r * ATS + cg], ga);
    if (BTRANS) {
      const unsigned short* gb = B + (size_t)(n0 + r) * ldb + bcol + k0 + cg;
      async_ld_b128x2(&Bt[buf][r * ATS + cg], gb);
    } else {
      // weights [K,N]: row kk contiguous along n; scatter into B^T
      union { uint4 q[2]; unsigned short s2[16]; } u;
      const unsigned short* gb = B + (size_t)(k0 + r) * ldb + bcol + n0 + cg;
      u.q[0] = ((const uint4*)gb)[0];
      u.q[1] = ((const uint4*)gb)[1];
#pragma unroll
      for (int i = 0; i < 16; ++i) Bt[buf][(cg + i) * ATS + r] = u.s2[i];
    }
  };

  load_stage(0, 0);
  for (int s = 0; s < S; ++s) {
    int cur = s & 1;
    __syncthreads();                       // everyone done reading buf cur^1
    int snxt = (s + 1 < S) ? (s + 1) : s;  // clamp: redundant reload, no branch
    load_stage(cur ^ 1, snxt << 6);
    wait_async<PEND>();                    // stage-s transfers of this wave done
    __syncthreads();                       // whole tile visible
    const unsigned int* arow = (const unsigned int*)&As[cur][(i0 * 16 + (lane & 15)) * ATS];
    const unsigned int* br0  = (const unsigned int*)&Bt[cur][((jb + 0) * 16 + (lane & 15)) * ATS];
    const unsigned int* br1  = (const unsigned int*)&Bt[cur][((jb + 1) * 16 + (lane & 15)) * ATS];
    bf16x16 a0 = load_fragA(arow, lane, 0);
    bf16x16 a1 = load_fragA(arow, lane, 16);
    acc0 = wmma_bf16(a0, load_fragB(br0, lane, 0),  acc0);
    acc0 = wmma_bf16(a1, load_fragB(br0, lane, 16), acc0);
    acc1 = wmma_bf16(a0, load_fragB(br1, lane, 0),  acc1);
    acc1 = wmma_bf16(a1, load_fragB(br1, lane, 16), acc1);
  }

  int halfadd = (lane >> 4) ? 8 : 0;
  int nc0 = n0 + jb * 16 + (lane & 15);
  int nc1 = nc0 + 16;
  float b0a = bias0 ? bias0[nc0] : 0.0f;
  float b0b = bias0 ? bias0[nc1] : 0.0f;
  float b1a = bias1 ? bias1[nc0] : 0.0f;
  float b1b = bias1 ? bias1[nc1] : 0.0f;
#pragma unroll
  for (int rr = 0; rr < 8; ++rr) {
    int mr = m0 + i0 * 16 + rr + halfadd;
    size_t idx0 = cbase + (size_t)mr * ldc + nc0;
    size_t idx1 = idx0 + 16;
    if (out_f32) {
      ((float*)C0)[idx0] = acc0[rr] + b0a;
      ((float*)C0)[idx1] = acc1[rr] + b0b;
    } else {
      ((unsigned short*)C0)[idx0] = f2bf(acc0[rr] + b0a);
      ((unsigned short*)C0)[idx1] = f2bf(acc1[rr] + b0b);
      if (C1) {
        ((unsigned short*)C1)[idx0] = f2bf(acc0[rr] + b1a);
        ((unsigned short*)C1)[idx1] = f2bf(acc1[rr] + b1b);
      }
    }
  }
}

// ----------------------- flash attention (one b, one h) --------------------
// Block = (q-tile of 64 rows) x (h). 8 waves; wave owns two 16x16 subtiles.
// Loops over 16 key tiles of 64; online softmax; pos score gathered from the
// precomputed PS[h][t][j] = (q+v_bias)_t . p_j via the exact rel_shift map.
__global__ void __launch_bounds__(256)
attn_kernel(const unsigned short* __restrict__ qu, const unsigned short* __restrict__ kmat,
            const unsigned short* __restrict__ vmat, const float* __restrict__ PSb,
            unsigned short* __restrict__ ctx, int b) {
  __shared__ unsigned short Qu[64 * ATS];
  __shared__ unsigned short Kt[64 * ATS];   // [s][dh]  (= B^T for QK^T)
  __shared__ unsigned short Vt[64 * ATS];   // [dh][s]  (= B^T for P.V)
  __shared__ unsigned short Pb[64 * ATS];   // exp(S-m) in bf16
  __shared__ float Sbuf[64 * 65];
  __shared__ float m_s[64], l_s[64], al_s[64];

  int tid = threadIdx.x, wave = tid >> 5, lane = tid & 31;
  int qt = blockIdx.x, h = blockIdx.y;
  int t0 = qt * 64;
  int i0 = wave >> 1;
  int jb = (wave & 1) * 2;
  int halfadd = (lane >> 4) ? 8 : 0;
  int r  = tid >> 2;          // 0..63
  int cg = (tid & 3) * 16;    // 0,16,32,48

  {  // Q(+u_bias) tile via async DMA: rows t0.., cols h*64..
    const unsigned short* g = qu + ((size_t)(b * DIMT + t0 + r)) * DIMD + h * DIMDH + cg;
    async_ld_b128x2(&Qu[r * ATS + cg], g);
  }
  if (tid < 64) { m_s[tid] = -1e30f; l_s[tid] = 0.0f; }
  f32x8 o0 = zero8(), o1 = zero8();

  for (int s0 = 0; s0 < DIMT; s0 += 64) {
    __syncthreads();  // previous iteration's LDS consumers are done
    {  // K tile via async DMA; V tile manual (transpose scatter into Vt)
      const unsigned short* gk = kmat + ((size_t)(b * DIMT + s0 + r)) * DIMD + h * DIMDH + cg;
      async_ld_b128x2(&Kt[r * ATS + cg], gk);
      union { uint4 q[2]; unsigned short us[16]; } uv;
      const uint4* gv = (const uint4*)(vmat + ((size_t)(b * DIMT + s0 + r)) * DIMD + h * DIMDH + cg);
      uv.q[0] = gv[0]; uv.q[1] = gv[1];
#pragma unroll
      for (int i = 0; i < 16; ++i) Vt[(cg + i) * ATS + r] = uv.us[i];
    }
    wait_async<0>();   // this wave's Qu/Kt DMAs have landed
    __syncthreads();

    // ---- S = (Q+u) . K^T  (K-dim = 64 -> 2 chained WMMAs per subtile) ----
    f32x8 sA = zero8(), sB = zero8();
    {
      const unsigned int* arow = (const unsigned int*)&Qu[(i0 * 16 + (lane & 15)) * ATS];
      const unsigned int* br0  = (const unsigned int*)&Kt[((jb + 0) * 16 + (lane & 15)) * ATS];
      const unsigned int* br1  = (const unsigned int*)&Kt[((jb + 1) * 16 + (lane & 15)) * ATS];
      bf16x16 a0 = load_fragA(arow, lane, 0);
      bf16x16 a1 = load_fragA(arow, lane, 16);
      sA = wmma_bf16(a0, load_fragB(br0, lane, 0),  sA);
      sA = wmma_bf16(a1, load_fragB(br0, lane, 16), sA);
      sB = wmma_bf16(a0, load_fragB(br1, lane, 0),  sB);
      sB = wmma_bf16(a1, load_fragB(br1, lane, 16), sB);
    }
#pragma unroll
    for (int rr = 0; rr < 8; ++rr) {
      Sbuf[(i0 * 16 + rr + halfadd) * 65 + (jb + 0) * 16 + (lane & 15)] = sA[rr];
      Sbuf[(i0 * 16 + rr + halfadd) * 65 + (jb + 1) * 16 + (lane & 15)] = sB[rr];
    }
    __syncthreads();

    // ---- add rel-shifted position score + 1/sqrt(D) scale ----
    {
      int tg = t0 + r;
#pragma unroll
      for (int i = 0; i < 16; ++i) {
        int sg = s0 + cg + i;
        unsigned int nn = (unsigned int)(tg + 1) * 1024u + (unsigned int)sg;
        unsigned int rS = nn / 1025u;
        unsigned int cS = nn % 1025u;
        float pos = 0.0f;
        if (cS != 0u) pos = PSb[((size_t)h * DIMT + rS) * DIMT + (cS - 1u)];
        Sbuf[r * 65 + cg + i] = (Sbuf[r * 65 + cg + i] + pos) * INV_SQRT_D;
      }
    }
    __syncthreads();

    // ---- online softmax over this 64-wide key tile ----
    if (tid < 64) {
      float mold = m_s[tid];
      float mt = -1e30f;
      for (int sI = 0; sI < 64; ++sI) mt = fmaxf(mt, Sbuf[tid * 65 + sI]);
      float mnew = fmaxf(mold, mt);
      float alpha = __expf(mold - mnew);
      float sum = 0.0f;
      for (int sI = 0; sI < 64; ++sI) {
        float p = __expf(Sbuf[tid * 65 + sI] - mnew);
        Pb[tid * ATS + sI] = f2bf(p);
        sum += p;
      }
      l_s[tid] = l_s[tid] * alpha + sum;
      m_s[tid] = mnew;
      al_s[tid] = alpha;
    }
    __syncthreads();

    // ---- O = alpha*O + P . V ----
    {
#pragma unroll
      for (int rr = 0; rr < 8; ++rr) {
        float a = al_s[i0 * 16 + rr + halfadd];
        o0[rr] *= a; o1[rr] *= a;
      }
      const unsigned int* prow = (const unsigned int*)&Pb[(i0 * 16 + (lane & 15)) * ATS];
      const unsigned int* vr0  = (const unsigned int*)&Vt[((jb + 0) * 16 + (lane & 15)) * ATS];
      const unsigned int* vr1  = (const unsigned int*)&Vt[((jb + 1) * 16 + (lane & 15)) * ATS];
      bf16x16 p0 = load_fragA(prow, lane, 0);
      bf16x16 p1 = load_fragA(prow, lane, 16);
      o0 = wmma_bf16(p0, load_fragB(vr0, lane, 0),  o0);
      o0 = wmma_bf16(p1, load_fragB(vr0, lane, 16), o0);
      o1 = wmma_bf16(p0, load_fragB(vr1, lane, 0),  o1);
      o1 = wmma_bf16(p1, load_fragB(vr1, lane, 16), o1);
    }
  }

  // ---- epilogue: ctx = O / l ----
#pragma unroll
  for (int rr = 0; rr < 8; ++rr) {
    int tl = i0 * 16 + rr + halfadd;
    float inv = 1.0f / l_s[tl];
    int col0 = h * DIMDH + jb * 16 + (lane & 15);
    size_t base = ((size_t)(b * DIMT + t0 + tl)) * DIMD;
    ctx[base + col0]      = f2bf(o0[rr] * inv);
    ctx[base + col0 + 16] = f2bf(o1[rr] * inv);
  }
}

// ---------------------------------------------------------------------------
extern "C" void kernel_launch(void* const* d_in, const int* in_sizes, int n_in,
                              void* d_out, int out_size, void* d_ws, size_t ws_size,
                              hipStream_t stream) {
  (void)in_sizes; (void)n_in; (void)out_size; (void)ws_size;
  const float* x    = (const float*)d_in[0];
  const float* ln_g = (const float*)d_in[1];
  const float* ln_b = (const float*)d_in[2];
  const float* Wq   = (const float*)d_in[3];
  const float* bq   = (const float*)d_in[4];
  const float* Wk   = (const float*)d_in[5];
  const float* bk   = (const float*)d_in[6];
  const float* Wv   = (const float*)d_in[7];
  const float* bv   = (const float*)d_in[8];
  const float* Wpos = (const float*)d_in[9];
  const float* ub   = (const float*)d_in[10];
  const float* vbias= (const float*)d_in[11];
  const float* Wo   = (const float*)d_in[12];
  const float* bo   = (const float*)d_in[13];
  float* out = (float*)d_out;

  const int BT = DIMB * DIMT;          // 8192 tokens
  size_t off = 0;
  char* wsb = (char*)d_ws;
  auto alloc = [&](size_t bytes) -> void* {
    void* p = (void*)(wsb + off);
    off += (bytes + 255) & ~(size_t)255;
    return p;
  };
  // ~89 MB of workspace (fits entirely in the 192MB global L2)
  unsigned short* d_xn  = (unsigned short*)alloc((size_t)BT * DIMD * 2);
  unsigned short* d_wq  = (unsigned short*)alloc((size_t)DIMD * DIMD * 2);
  unsigned short* d_wk  = (unsigned short*)alloc((size_t)DIMD * DIMD * 2);
  unsigned short* d_wv  = (unsigned short*)alloc((size_t)DIMD * DIMD * 2);
  unsigned short* d_wp  = (unsigned short*)alloc((size_t)DIMD * DIMD * 2);
  unsigned short* d_wo  = (unsigned short*)alloc((size_t)DIMD * DIMD * 2);
  unsigned short* d_pe  = (unsigned short*)alloc((size_t)DIMT * DIMD * 2);
  unsigned short* d_p   = (unsigned short*)alloc((size_t)DIMT * DIMD * 2);
  unsigned short* d_qu  = (unsigned short*)alloc((size_t)BT * DIMD * 2);
  unsigned short* d_qv  = (unsigned short*)alloc((size_t)BT * DIMD * 2);
  unsigned short* d_k   = (unsigned short*)alloc((size_t)BT * DIMD * 2);
  unsigned short* d_v   = (unsigned short*)alloc((size_t)BT * DIMD * 2);
  unsigned short* d_ctx = (unsigned short*)alloc((size_t)BT * DIMD * 2);
  float* d_bqu = (float*)alloc((size_t)DIMD * 4);
  float* d_bqv = (float*)alloc((size_t)DIMD * 4);
  float* d_ps  = (float*)alloc((size_t)DIMH * DIMT * DIMT * 4);  // per-batch, reused

  // 1) LayerNorm -> bf16 xn
  ln_kernel<<<dim3(BT), dim3(128), 0, stream>>>(x, ln_g, ln_b, d_xn);
  // 2) weights -> bf16
  const int WN = DIMD * DIMD;
  cvt_kernel<<<dim3(WN / 256), dim3(256), 0, stream>>>(Wq,   d_wq, WN);
  cvt_kernel<<<dim3(WN / 256), dim3(256), 0, stream>>>(Wk,   d_wk, WN);
  cvt_kernel<<<dim3(WN / 256), dim3(256), 0, stream>>>(Wv,   d_wv, WN);
  cvt_kernel<<<dim3(WN / 256), dim3(256), 0, stream>>>(Wpos, d_wp, WN);
  cvt_kernel<<<dim3(WN / 256), dim3(256), 0, stream>>>(Wo,   d_wo, WN);
  // 3) positional encoding (bf16) and fused biases
  pe_kernel<<<dim3((DIMT * DIMD) / 256), dim3(256), 0, stream>>>(d_pe);
  biasprep_kernel<<<dim3(2), dim3(256), 0, stream>>>(bq, ub, vbias, d_bqu, d_bqv);

  // 4) projections (bf16 WMMA GEMMs, weight-B path)
  dim3 gproj(DIMD / 64, BT / 64, 1);
  //   qu = xn@Wq + (bq+u) ; qv = xn@Wq + (bq+v)   (two-output epilogue)
  gemm_wmma_bf16<0><<<gproj, 256, 0, stream>>>(d_xn, DIMD, 0, 0, d_wq, DIMD, 0, 0,
                                               (void*)d_qu, (void*)d_qv, 0, DIMD, 0,
                                               d_bqu, d_bqv, BT, DIMD, DIMD);
  gemm_wmma_bf16<0><<<gproj, 256, 0, stream>>>(d_xn, DIMD, 0, 0, d_wk, DIMD, 0, 0,
                                               (void*)d_k, nullptr, 0, DIMD, 0,
                                               bk, nullptr, BT, DIMD, DIMD);
  gemm_wmma_bf16<0><<<gproj, 256, 0, stream>>>(d_xn, DIMD, 0, 0, d_wv, DIMD, 0, 0,
                                               (void*)d_v, nullptr, 0, DIMD, 0,
                                               bv, nullptr, BT, DIMD, DIMD);
  dim3 gpos(DIMD / 64, DIMT / 64, 1);
  gemm_wmma_bf16<0><<<gpos, 256, 0, stream>>>(d_pe, DIMD, 0, 0, d_wp, DIMD, 0, 0,
                                              (void*)d_p, nullptr, 0, DIMD, 0,
                                              nullptr, nullptr, DIMT, DIMD, DIMD);

  // 5) per-batch: PS[h][t][j] = (q+v)_t . p_j  (B-transposed async path),
  //    then flash attention for that batch
  for (int b = 0; b < DIMB; ++b) {
    dim3 gps(DIMT / 64, DIMT / 64, DIMH);   // z = head slice (za=zb=64)
    gemm_wmma_bf16<1><<<gps, 256, 0, stream>>>(
        d_qv + (size_t)b * DIMT * DIMD, DIMD, 0, DIMDH,
        d_p, DIMD, 0, DIMDH,
        (void*)d_ps, nullptr, /*out_f32=*/1, DIMT, (long long)DIMT * DIMT,
        nullptr, nullptr, DIMT, DIMT, DIMDH);
    dim3 gat(DIMT / 64, DIMH, 1);
    attn_kernel<<<gat, 256, 0, stream>>>(d_qu, d_k, d_v, d_ps, d_ctx, b);
  }

  // 6) output projection -> fp32 d_out
  gemm_wmma_bf16<0><<<gproj, 256, 0, stream>>>(d_ctx, DIMD, 0, 0, d_wo, DIMD, 0, 0,
                                               (void*)out, nullptr, /*out_f32=*/1, DIMD, 0,
                                               bo, nullptr, BT, DIMD, DIMD);
}
